// KGCN_Aggregate_4629974745848
// MI455X (gfx1250) — compile-verified, matrix operand order checked by reading, not run on hardware
//
#include <hip/hip_runtime.h>
#include <hip/hip_bf16.h>
#include <math.h>

typedef __attribute__((ext_vector_type(16))) _Float16 v16h;
typedef __attribute__((ext_vector_type(8)))  float    v8f;

#define DFEAT 64

// ---------------------------------------------------------------------------
// Phase 0: h = emb_dst  (accumulator lives in d_ws, 100000*64 floats = 25.6MB)
// ---------------------------------------------------------------------------
__global__ void kgcn_init_h(const float* __restrict__ emb_dst,
                            float* __restrict__ h, int total) {
    int i = blockIdx.x * blockDim.x + threadIdx.x;
    if (i < total) h[i] = emb_dst[i];
}

// ---------------------------------------------------------------------------
// Phase 1: scatter-add  h[dst] += emb_src[src] * w  (L2-resident atomics)
// 64 consecutive lanes = one edge row -> coalesced loads + coalesced atomics.
// ---------------------------------------------------------------------------
__global__ void kgcn_scatter(const float* __restrict__ emb_src,
                             const float* __restrict__ ew,
                             const int*   __restrict__ src_idx,
                             const int*   __restrict__ dst_idx,
                             float* __restrict__ h, int E) {
    int tid = blockIdx.x * blockDim.x + threadIdx.x;
    int e = tid >> 6;
    int f = tid & 63;
    if (e < E) {
        int s = src_idx[e];
        int d = dst_idx[e];
        float w = ew[e];
        float v = emb_src[(size_t)s * DFEAT + f] * w;
        unsafeAtomicAdd(&h[(size_t)d * DFEAT + f], v);  // global_atomic_add_f32
    }
}

// K-index map for the CDNA5 16-bit A 16x32 fragment (and symmetric B 32x16):
// per lane (hi = lane>>4): v16h element j -> VGPR v=j/2, sub=j%2
//   v in 0..3 : K = 2v       + sub + 8*hi
//   v in 4..7 : K = 16+2(v-4)+ sub + 8*hi
__device__ __forceinline__ int wmma_kmap(int j, int hi) {
    int v = j >> 1, sub = j & 1;
    int base = (v < 4) ? (2 * v) : (16 + 2 * (v - 4));
    return base + sub + 8 * hi;
}

// ---------------------------------------------------------------------------
// Phase 2: out = tanh(h @ W + b) via v_wmma_f32_16x16x32_f16, K=64 in 2 steps.
// One wave per 16x16 output tile; 8 waves (tiles) per 256-thread block.
// 25000 tiles total (100000/16 row-tiles * 4 col-tiles) -> 3125 blocks exact,
// so every wave is fully active (EXEC all-1s, required by WMMA).
// ---------------------------------------------------------------------------
__global__ void __launch_bounds__(256)
kgcn_gemm_tanh(const float* __restrict__ h,
               const float* __restrict__ W,
               const float* __restrict__ bias,
               float* __restrict__ out) {
    int wave = threadIdx.x >> 5;
    int lane = threadIdx.x & 31;
    int tile = blockIdx.x * 8 + wave;
    int row0 = (tile >> 2) * 16;   // node tile
    int col0 = (tile & 3) * 16;    // output-feature tile
    int m  = lane & 15;
    int hi = lane >> 4;

    const float* hrow = h + (size_t)(row0 + m) * DFEAT;

    v16h a0, a1, b0, b1;
#pragma unroll
    for (int j = 0; j < 16; ++j) {
        int k = wmma_kmap(j, hi);
        a0[j] = (_Float16)hrow[k];
        a1[j] = (_Float16)hrow[k + 32];
        b0[j] = (_Float16)W[(size_t)k        * DFEAT + (col0 + m)];
        b1[j] = (_Float16)W[(size_t)(k + 32) * DFEAT + (col0 + m)];
    }

    v8f c = {};
    c = __builtin_amdgcn_wmma_f32_16x16x32_f16(false, a0, false, b0,
                                               (short)0, c, false, false);
    c = __builtin_amdgcn_wmma_f32_16x16x32_f16(false, a1, false, b1,
                                               (short)0, c, false, false);

    float bb = bias[col0 + m];
#pragma unroll
    for (int r = 0; r < 8; ++r) {
        int M = r + 8 * hi;   // C/D layout: VGPR r, lanes16-31 -> M+8
        out[(size_t)(row0 + M) * DFEAT + col0 + m] = tanhf(c[r] + bb);
    }
}

// ---------------------------------------------------------------------------
// Phase 3: users = user_indices[perm], labs = labels[perm] (raw int32 copy)
// ---------------------------------------------------------------------------
__global__ void kgcn_gather_i32(const int* __restrict__ vals,
                                const int* __restrict__ perm,
                                int* __restrict__ out, int n) {
    int i = blockIdx.x * blockDim.x + threadIdx.x;
    if (i < n) out[i] = vals[perm[i]];
}

extern "C" void kernel_launch(void* const* d_in, const int* in_sizes, int n_in,
                              void* d_out, int out_size, void* d_ws, size_t ws_size,
                              hipStream_t stream) {
    const float* emb_src  = (const float*)d_in[0];
    const float* emb_dst  = (const float*)d_in[1];
    const float* ew       = (const float*)d_in[2];
    const float* W        = (const float*)d_in[3];
    const float* bias     = (const float*)d_in[4];
    const int*   src_idx  = (const int*)  d_in[5];
    const int*   dst_idx  = (const int*)  d_in[6];
    const int*   user_idx = (const int*)  d_in[7];
    const int*   labels   = (const int*)  d_in[8];
    const int*   perm     = (const int*)  d_in[9];

    const int E = in_sizes[2];     // 1,600,000
    const int N = in_sizes[7];     // 100,000

    float* h        = (float*)d_ws;                     // [N,64] accumulator
    float* out_emb  = (float*)d_out;                    // [N,64] f32
    int*   out_user = (int*)d_out + (size_t)N * DFEAT;  // [N] i32 (raw bits)
    int*   out_labs = out_user + N;                     // [N] i32 (raw bits)

    const int totalHD = N * DFEAT;
    kgcn_init_h<<<(totalHD + 255) / 256, 256, 0, stream>>>(emb_dst, h, totalHD);

    const long long totE = (long long)E * DFEAT;        // 102.4M lane-items
    kgcn_scatter<<<(int)((totE + 255) / 256), 256, 0, stream>>>(
        emb_src, ew, src_idx, dst_idx, h, E);

    const int tiles = (N / 16) * 4;                     // 25000, divisible by 8
    kgcn_gemm_tanh<<<tiles / 8, 256, 0, stream>>>(h, W, bias, out_emb);

    kgcn_gather_i32<<<(N + 255) / 256, 256, 0, stream>>>(user_idx, perm, out_user, N);
    kgcn_gather_i32<<<(N + 255) / 256, 256, 0, stream>>>(labels,   perm, out_labs, N);
}
